// DeepRegParm_25701084299685
// MI455X (gfx1250) — compile-verified, matrix operand order checked by reading, not run on hardware
//
#include <hip/hip_runtime.h>
#include <cstddef>

typedef __attribute__((ext_vector_type(16))) _Float16 v16h;
typedef __attribute__((ext_vector_type(8)))  float    v8f;
typedef _Float16 h16;

// ------------------------------------------------------------------
// Kernels
// ------------------------------------------------------------------

// Farthest point sampling: one 256-thread block per batch.
__global__ void k_fps(const float* __restrict__ xyz, int* __restrict__ fidx,
                      float* __restrict__ dist, int N, int npoint)
{
  int b = blockIdx.x;
  const float* pts = xyz + (size_t)b * N * 3;
  float* db = dist + (size_t)b * N;
  int*   out = fidx + (size_t)b * npoint;
  __shared__ float sval[256];
  __shared__ int   sidx[256];
  int tid = threadIdx.x;
  for (int i = tid; i < N; i += 256) db[i] = 1e10f;
  __syncthreads();
  int far = 0;
  for (int it = 0; it < npoint; ++it) {
    if (tid == 0) out[it] = far;
    float cx = pts[3*far+0], cy = pts[3*far+1], cz = pts[3*far+2];
    float best = -1.0f; int bi = 0;
    for (int i = tid; i < N; i += 256) {
      float dx = pts[3*i+0]-cx, dy = pts[3*i+1]-cy, dz = pts[3*i+2]-cz;
      float d  = dx*dx + dy*dy + dz*dz;
      float nd = fminf(db[i], d);
      db[i] = nd;
      if (nd > best) { best = nd; bi = i; }
    }
    sval[tid] = best; sidx[tid] = bi;
    __syncthreads();
    for (int s = 128; s > 0; s >>= 1) {
      if (tid < s) {
        float ov = sval[tid+s]; int oi = sidx[tid+s];
        if (ov > sval[tid] || (ov == sval[tid] && oi < sidx[tid])) { sval[tid] = ov; sidx[tid] = oi; }
      }
      __syncthreads();
    }
    far = sidx[0];
    __syncthreads();
  }
}

__global__ void k_gather3(const float* __restrict__ xyz, const int* __restrict__ idx,
                          float* __restrict__ out, int B, int N, int M)
{
  int t = blockIdx.x * blockDim.x + threadIdx.x;
  if (t >= B * M) return;
  int b = t / M;
  int id = idx[t];
  const float* p = xyz + ((size_t)b * N + id) * 3;
  out[(size_t)t*3+0] = p[0]; out[(size_t)t*3+1] = p[1]; out[(size_t)t*3+2] = p[2];
}

// Ball query: sorted-index semantics == first nsample in-radius indices.
__global__ void k_ball_query(const float* __restrict__ xyz, const float* __restrict__ nxyz,
                             int* __restrict__ gidx, int B, int N, int S, int ns, float r2)
{
  int t = blockIdx.x * blockDim.x + threadIdx.x;
  if (t >= B * S) return;
  int b = t / S;
  const float* q = nxyz + (size_t)t * 3;
  float qx = q[0], qy = q[1], qz = q[2];
  const float* pts = xyz + (size_t)b * N * 3;
  int* out = gidx + (size_t)t * ns;
  int cnt = 0;
  for (int j = 0; j < N && cnt < ns; ++j) {
    float dx = pts[3*j+0]-qx, dy = pts[3*j+1]-qy, dz = pts[3*j+2]-qz;
    float d = dx*dx + dy*dy + dz*dz;
    if (d <= r2) out[cnt++] = j;
  }
  int first = (cnt > 0) ? out[0] : 0;
  for (int c = cnt; c < ns; ++c) out[c] = first;
}

// kNN (top-K ascending distance) via insertion sort.
template <int K>
__global__ void k_knn(const float* __restrict__ ref, const float* __restrict__ query,
                      int* __restrict__ idx, int B, int Nr, int Nq)
{
  int t = blockIdx.x * blockDim.x + threadIdx.x;
  if (t >= B * Nq) return;
  int b = t / Nq;
  const float* q = query + (size_t)t * 3;
  float qx = q[0], qy = q[1], qz = q[2];
  float bd[K]; int bi[K];
  for (int k = 0; k < K; ++k) { bd[k] = 3.0e38f; bi[k] = 0; }
  const float* rp = ref + (size_t)b * Nr * 3;
  for (int j = 0; j < Nr; ++j) {
    float dx = rp[3*j+0]-qx, dy = rp[3*j+1]-qy, dz = rp[3*j+2]-qz;
    float d = dx*dx + dy*dy + dz*dz;
    if (d < bd[K-1]) {
      int p = K - 1;
      while (p > 0 && bd[p-1] > d) { bd[p] = bd[p-1]; bi[p] = bi[p-1]; --p; }
      bd[p] = d; bi[p] = j;
    }
  }
  int* o = idx + (size_t)t * K;
  for (int k = 0; k < K; ++k) o[k] = bi[k];
}

// Grouping kernels: build padded f16 activation rows for WMMA.
__global__ void k_group_sa(const float* __restrict__ xyz, const float* __restrict__ feats,
                           const float* __restrict__ nxyz, const int* __restrict__ gidx,
                           h16* __restrict__ Xh, int B, int N, int S, int ns, int C, int kpad)
{
  int t = blockIdx.x * blockDim.x + threadIdx.x;
  int total = B * S * ns;
  if (t >= total) return;
  int s = (t / ns) % S;
  int b = t / (ns * S);
  int id = gidx[t];
  const float* p = xyz  + ((size_t)b * N + id) * 3;
  const float* q = nxyz + ((size_t)b * S + s) * 3;
  h16* o = Xh + (size_t)t * kpad;
  o[0] = (h16)(p[0]-q[0]); o[1] = (h16)(p[1]-q[1]); o[2] = (h16)(p[2]-q[2]);
  const float* f = feats + ((size_t)b * N + id) * C;
  for (int c = 0; c < C; ++c) o[3+c] = (h16)f[c];
  for (int c = 3 + C; c < kpad; ++c) o[c] = (h16)0.0f;
}

__global__ void k_group_fe(const float* __restrict__ pos1, const float* __restrict__ pos2,
                           const float* __restrict__ f1, const float* __restrict__ f2,
                           const int* __restrict__ idx, h16* __restrict__ Xh,
                           int B, int Nq, int Nr, int C1, int C2, int ns, int kpad)
{
  int t = blockIdx.x * blockDim.x + threadIdx.x;
  int total = B * Nq * ns;
  if (t >= total) return;
  int s = (t / ns) % Nq;
  int b = t / (ns * Nq);
  int id = idx[t];
  const float* p2 = pos2 + ((size_t)b * Nr + id) * 3;
  const float* p1 = pos1 + ((size_t)b * Nq + s) * 3;
  h16* o = Xh + (size_t)t * kpad;
  o[0] = (h16)(p2[0]-p1[0]); o[1] = (h16)(p2[1]-p1[1]); o[2] = (h16)(p2[2]-p1[2]);
  const float* g2 = f2 + ((size_t)b * Nr + id) * C2;
  for (int c = 0; c < C2; ++c) o[3+c] = (h16)g2[c];
  const float* g1 = f1 + ((size_t)b * Nq + s) * C1;
  for (int c = 0; c < C1; ++c) o[3+C2+c] = (h16)g1[c];
  for (int c = 3 + C2 + C1; c < kpad; ++c) o[c] = (h16)0.0f;
}

__global__ void k_group_up(const float* __restrict__ pos1, const float* __restrict__ pos2,
                           const float* __restrict__ f2, const int* __restrict__ idx,
                           h16* __restrict__ Xh, int B, int N1, int N2, int C2, int ns, int kpad)
{
  int t = blockIdx.x * blockDim.x + threadIdx.x;
  int total = B * N1 * ns;
  if (t >= total) return;
  int s = (t / ns) % N1;
  int b = t / (ns * N1);
  int id = idx[t];
  const float* p2 = pos2 + ((size_t)b * N2 + id) * 3;
  const float* p1 = pos1 + ((size_t)b * N1 + s) * 3;
  h16* o = Xh + (size_t)t * kpad;
  const float* g2 = f2 + ((size_t)b * N2 + id) * C2;
  for (int c = 0; c < C2; ++c) o[c] = (h16)g2[c];
  o[C2+0] = (h16)(p2[0]-p1[0]); o[C2+1] = (h16)(p2[1]-p1[1]); o[C2+2] = (h16)(p2[2]-p1[2]);
  for (int c = C2 + 3; c < kpad; ++c) o[c] = (h16)0.0f;
}

// Weight conversion fp32[cout,cin] -> f16[coutPad, kpad] zero-padded.
__global__ void k_convert_w(const float* __restrict__ W, h16* __restrict__ Wh,
                            int cout, int cin, int kpad, int total)
{
  int t = blockIdx.x * blockDim.x + threadIdx.x;
  if (t >= total) return;
  int k = t % kpad;
  int o = t / kpad;
  Wh[t] = (o < cout && k < cin) ? (h16)W[(size_t)o * cin + k] : (h16)0.0f;
}

// WMMA GEMM: Y[rows,cout] = X[rows,kpad] * Wt[coutPad,kpad]^T.
// Register-blocked: one wave computes 16 rows x (16*NT) cols; the A fragment
// is loaded once per K-step and reused for NT WMMAs (B = weights, L2-resident).
// 4 waves per block along M.
template <int NT>
__global__ void k_gemm_wmma(const h16* __restrict__ X, const h16* __restrict__ Wt,
                            float* __restrict__ Y, int rows, int kpad, int cout)
{
  int wave = threadIdx.x >> 5;
  int lane = threadIdx.x & 31;
  int half = lane >> 4;
  int l15  = lane & 15;
  int tm  = (blockIdx.x * 4 + wave) * 16;
  int tn0 = blockIdx.y * 16 * NT;
  int mr  = tm + l15;
  int mrc = mr < rows ? mr : rows - 1;   // keep loads in-bounds; store is guarded
  const h16* xaBase = X + (size_t)mrc * kpad;
  const h16* wbBase[NT];
#pragma unroll
  for (int t = 0; t < NT; ++t) wbBase[t] = Wt + (size_t)(tn0 + t * 16 + l15) * kpad;
  v8f acc[NT];
  v8f zero = {};
#pragma unroll
  for (int t = 0; t < NT; ++t) acc[t] = zero;
  for (int k0 = 0; k0 < kpad; k0 += 32) {
    if (k0 + 32 < kpad) __builtin_prefetch(xaBase + k0 + 32, 0, 1);
    const h16* xa = xaBase + k0 + half * 8;
    v16h a;
#pragma unroll
    for (int i = 0; i < 8; ++i) { a[i] = xa[i]; a[8 + i] = xa[16 + i]; }
#pragma unroll
    for (int t = 0; t < NT; ++t) {
      const h16* wb = wbBase[t] + k0 + half * 8;
      v16h bf;
#pragma unroll
      for (int i = 0; i < 8; ++i) { bf[i] = wb[i]; bf[8 + i] = wb[16 + i]; }
      acc[t] = __builtin_amdgcn_wmma_f32_16x16x32_f16(false, a, false, bf, (short)0, acc[t], false, false);
    }
  }
#pragma unroll
  for (int t = 0; t < NT; ++t) {
    int nc = tn0 + t * 16 + l15;
    if (nc < cout) {
#pragma unroll
      for (int i = 0; i < 8; ++i) {
        int m = tm + (half ? i + 8 : i);
        if (m < rows) Y[(size_t)m * cout + nc] = acc[t][i];
      }
    }
  }
}

__global__ void k_zero(float* __restrict__ p, int n)
{
  for (int i = threadIdx.x; i < n; i += blockDim.x) p[i] = 0.0f;
}

// Per-channel sum/sumsq reduction with LDS atomics then global atomics.
__global__ void k_colreduce(const float* __restrict__ Y, long total, int C,
                            float* __restrict__ sum, float* __restrict__ ssq)
{
  __shared__ float ls[256];
  __shared__ float lq[256];
  for (int c = threadIdx.x; c < C; c += blockDim.x) { ls[c] = 0.0f; lq[c] = 0.0f; }
  __syncthreads();
  long stride = (long)gridDim.x * blockDim.x;
  for (long i = (long)blockIdx.x * blockDim.x + threadIdx.x; i < total; i += stride) {
    float v = Y[i];
    int c = (int)(i % C);
    atomicAdd(&ls[c], v);
    atomicAdd(&lq[c], v * v);
  }
  __syncthreads();
  for (int c = threadIdx.x; c < C; c += blockDim.x) {
    atomicAdd(&sum[c], ls[c]);
    atomicAdd(&ssq[c], lq[c]);
  }
}

// BN + ReLU epilogue; writes fp32 result and padded f16 for the next WMMA layer.
__global__ void k_bn_relu(const float* __restrict__ Yin, float* __restrict__ Yout,
                          h16* __restrict__ Hout, int rows, int C, int kpadOut,
                          const float* __restrict__ g, const float* __restrict__ bt,
                          const float* __restrict__ sum, const float* __restrict__ ssq, float invR)
{
  int r = blockIdx.x * blockDim.x + threadIdx.x;
  if (r >= rows) return;
  const float* yi = Yin + (size_t)r * C;
  float* yo = Yout + (size_t)r * C;
  h16* ho = Hout + (size_t)r * kpadOut;
  for (int c = 0; c < C; ++c) {
    float mu  = sum[c] * invR;
    float var = ssq[c] * invR - mu * mu;
    float v = (yi[c] - mu) * rsqrtf(var + 1e-5f) * g[c] + bt[c];
    v = fmaxf(v, 0.0f);
    yo[c] = v;
    ho[c] = (h16)v;
  }
  for (int c = C; c < kpadOut; ++c) ho[c] = (h16)0.0f;
}

__global__ void k_maxpool_f(const float* __restrict__ Y, float* __restrict__ out,
                            int P, int ns, int C)
{
  int t = blockIdx.x * blockDim.x + threadIdx.x;
  if (t >= P * C) return;
  int c = t % C, p = t / C;
  const float* y = Y + ((size_t)p * ns) * C + c;
  float m = -3.0e38f;
  for (int j = 0; j < ns; ++j) m = fmaxf(m, y[(size_t)j * C]);
  out[(size_t)p * C + c] = m;
}

__global__ void k_maxpool_h(const h16* __restrict__ X, h16* __restrict__ out,
                            int P, int ns, int C, int kpadIn, int kpadOut, int chOff)
{
  int t = blockIdx.x * blockDim.x + threadIdx.x;
  if (t >= P * C) return;
  int c = t % C, p = t / C;
  const h16* x = X + ((size_t)p * ns) * kpadIn + c;
  float m = -3.0e38f;
  for (int j = 0; j < ns; ++j) m = fmaxf(m, (float)x[(size_t)j * kpadIn]);
  out[(size_t)p * kpadOut + chOff + c] = (h16)m;
}

__global__ void k_fill_h(h16* __restrict__ p, long n)
{
  long t = (long)blockIdx.x * blockDim.x + threadIdx.x;
  if (t < n) p[t] = (h16)0.0f;
}

__global__ void k_copyfeat_h(const float* __restrict__ src, h16* __restrict__ dst,
                             int P, int C, int kpad, int off)
{
  int t = blockIdx.x * blockDim.x + threadIdx.x;
  if (t >= P * C) return;
  int c = t % C, p = t / C;
  dst[(size_t)p * kpad + off + c] = (h16)src[(size_t)p * C + c];
}

__global__ void k_copy_cols_f(const float* __restrict__ src, float* __restrict__ dst,
                              int P, int Cs, int Cd, int off)
{
  int t = blockIdx.x * blockDim.x + threadIdx.x;
  if (t >= P * Cs) return;
  int c = t % Cs, p = t / Cs;
  dst[(size_t)p * Cd + off + c] = src[(size_t)p * Cs + c];
}

// Feature propagation: 3-NN inverse-distance interp + concat weight channel.
__global__ void k_fp_interp(const float* __restrict__ pos1, const float* __restrict__ pos2,
                            const float* __restrict__ f1, const float* __restrict__ f2,
                            h16* __restrict__ Xh, int B, int N1, int N2, int C2, int kpad)
{
  int t = blockIdx.x * blockDim.x + threadIdx.x;
  if (t >= B * N1) return;
  int b = t / N1;
  const float* q = pos1 + (size_t)t * 3;
  float qx = q[0], qy = q[1], qz = q[2];
  float d0 = 3.0e38f, d1 = 3.0e38f, d2 = 3.0e38f;
  int i0 = 0, i1 = 0, i2 = 0;
  const float* rp = pos2 + (size_t)b * N2 * 3;
  for (int j = 0; j < N2; ++j) {
    float dx = rp[3*j+0]-qx, dy = rp[3*j+1]-qy, dz = rp[3*j+2]-qz;
    float d = dx*dx + dy*dy + dz*dz;
    if (d < d0)      { d2=d1; i2=i1; d1=d0; i1=i0; d0=d; i0=j; }
    else if (d < d1) { d2=d1; i2=i1; d1=d;  i1=j; }
    else if (d < d2) { d2=d;  i2=j; }
  }
  float w0 = 1.0f / (fmaxf(d0, 0.0f) + 1e-8f);
  float w1 = 1.0f / (fmaxf(d1, 0.0f) + 1e-8f);
  float w2 = 1.0f / (fmaxf(d2, 0.0f) + 1e-8f);
  float ws = w0 + w1 + w2;
  w0 /= ws; w1 /= ws; w2 /= ws;
  const float* fa = f2 + ((size_t)b * N2 + i0) * C2;
  const float* fb = f2 + ((size_t)b * N2 + i1) * C2;
  const float* fc = f2 + ((size_t)b * N2 + i2) * C2;
  h16* o = Xh + (size_t)t * kpad;
  for (int c = 0; c < C2; ++c) o[c] = (h16)(w0*fa[c] + w1*fb[c] + w2*fc[c]);
  o[C2] = (h16)f1[t];
  for (int c = C2 + 1; c < kpad; ++c) o[c] = (h16)0.0f;
}

__global__ void k_bias3(float* __restrict__ Y, const float* __restrict__ bias, int rows)
{
  int t = blockIdx.x * blockDim.x + threadIdx.x;
  if (t >= rows * 3) return;
  Y[t] += bias[t % 3];
}

// ------------------------------------------------------------------
// Host orchestration
// ------------------------------------------------------------------

static inline int pad32i(int x) { return (x + 31) & ~31; }
static inline int pad16i(int x) { return (x + 15) & ~15; }

struct Bump {
  char* base; size_t off;
  void* alloc(size_t n) {
    size_t a = (off + 255) & ~(size_t)255;
    off = a + n;
    return (void*)(base + a);
  }
};

struct Layer { const float* W; const float* g; const float* b; int cin; int cout; };

static void launch_gemm(hipStream_t st, const h16* X, const h16* Wh, float* Y,
                        int rows, int kpad, int coutPad, int cout)
{
  dim3 gg((rows + 63) / 64, 1);
  if (coutPad % 64 == 0) {
    gg.y = coutPad / 64;
    k_gemm_wmma<4><<<gg, 128, 0, st>>>(X, Wh, Y, rows, kpad, cout);
  } else if (coutPad % 32 == 0) {
    gg.y = coutPad / 32;
    k_gemm_wmma<2><<<gg, 128, 0, st>>>(X, Wh, Y, rows, kpad, cout);
  } else {
    gg.y = coutPad / 16;
    k_gemm_wmma<1><<<gg, 128, 0, st>>>(X, Wh, Y, rows, kpad, cout);
  }
}

static float* run_mlp(hipStream_t st, Bump& sb, const h16* X0, int rows, int kpad0,
                      const Layer* Ls, int nL, float* Yfinal, const h16** XhFinal)
{
  const h16* x = X0;
  int kpad = kpad0;
  float* Y = nullptr;
  for (int l = 0; l < nL; ++l) {
    const Layer& L = Ls[l];
    int coutPad = pad16i(L.cout);
    h16* Wh = (h16*)sb.alloc((size_t)coutPad * kpad * sizeof(h16));
    int wtotal = coutPad * kpad;
    k_convert_w<<<dim3((wtotal + 255) / 256), 256, 0, st>>>(L.W, Wh, L.cout, L.cin, kpad, wtotal);
    bool last = (l == nL - 1);
    float* Yl = (last && Yfinal) ? Yfinal : (float*)sb.alloc((size_t)rows * L.cout * sizeof(float));
    launch_gemm(st, x, Wh, Yl, rows, kpad, coutPad, L.cout);
    float* sums = (float*)sb.alloc(2 * (size_t)L.cout * sizeof(float));
    k_zero<<<1, 256, 0, st>>>(sums, 2 * L.cout);
    long total = (long)rows * L.cout;
    int blocks = (int)((total + 255) / 256); if (blocks > 2048) blocks = 2048;
    k_colreduce<<<blocks, 256, 0, st>>>(Yl, total, L.cout, sums, sums + L.cout);
    int kpadN = pad32i(L.cout);
    h16* Xn = (h16*)sb.alloc((size_t)rows * kpadN * sizeof(h16));
    k_bn_relu<<<dim3((rows + 255) / 256), 256, 0, st>>>(Yl, Yl, Xn, rows, L.cout, kpadN,
                                                        L.g, L.b, sums, sums + L.cout, 1.0f / rows);
    x = Xn; kpad = kpadN; Y = Yl;
  }
  if (XhFinal) *XhFinal = x;
  return Y;
}

static void sa_host(hipStream_t st, Bump& sb, size_t mark,
                    const float* xyz, const float* feats, int B, int Nin, int C,
                    int npoint, float radius, int ns, const Layer* Ls, int nL,
                    float* out_xyz, float* out_feats)
{
  sb.off = mark;
  int* fidx = (int*)sb.alloc((size_t)B * npoint * sizeof(int));
  float* dist = (float*)sb.alloc((size_t)B * Nin * sizeof(float));
  k_fps<<<B, 256, 0, st>>>(xyz, fidx, dist, Nin, npoint);
  k_gather3<<<dim3((B * npoint + 255) / 256), 256, 0, st>>>(xyz, fidx, out_xyz, B, Nin, npoint);
  int* gidx = (int*)sb.alloc((size_t)B * npoint * ns * sizeof(int));
  k_ball_query<<<dim3((B * npoint + 127) / 128), 128, 0, st>>>(xyz, out_xyz, gidx, B, Nin, npoint, ns,
                                                               radius * radius);
  int rows = B * npoint * ns;
  int kpad = pad32i(3 + C);
  h16* Xh = (h16*)sb.alloc((size_t)rows * kpad * sizeof(h16));
  k_group_sa<<<dim3((rows + 255) / 256), 256, 0, st>>>(xyz, feats, out_xyz, gidx, Xh, B, Nin, npoint, ns, C, kpad);
  float* Yf = run_mlp(st, sb, Xh, rows, kpad, Ls, nL, nullptr, nullptr);
  int Cl = Ls[nL - 1].cout;
  k_maxpool_f<<<dim3((B * npoint * Cl + 255) / 256), 256, 0, st>>>(Yf, out_feats, B * npoint, ns, Cl);
}

static void fe_host(hipStream_t st, Bump& sb, size_t mark,
                    const float* pos1, const float* pos2, const float* f1, const float* f2,
                    int B, int Npt, int C, const Layer* Ls, int nL, float* out)
{
  sb.off = mark;
  const int K = 64;
  int* idx = (int*)sb.alloc((size_t)B * Npt * K * sizeof(int));
  k_knn<64><<<dim3((B * Npt + 63) / 64), 64, 0, st>>>(pos2, pos1, idx, B, Npt, Npt);
  int rows = B * Npt * K;
  int kpad = pad32i(3 + 2 * C);
  h16* Xh = (h16*)sb.alloc((size_t)rows * kpad * sizeof(h16));
  k_group_fe<<<dim3((rows + 255) / 256), 256, 0, st>>>(pos1, pos2, f1, f2, idx, Xh, B, Npt, Npt, C, C, K, kpad);
  float* Yf = run_mlp(st, sb, Xh, rows, kpad, Ls, nL, nullptr, nullptr);
  int Cl = Ls[nL - 1].cout;
  k_maxpool_f<<<dim3((B * Npt * Cl + 255) / 256), 256, 0, st>>>(Yf, out, B * Npt, K, Cl);
}

static void upconv_host(hipStream_t st, Bump& sb, size_t mark,
                        const float* pos1, const float* pos2, const float* f1, const float* f2,
                        int B, int N1, int N2, int C1, int C2, int ns,
                        const Layer* mlp, int nMlp, const Layer* mlp2, int nMlp2, float* out)
{
  sb.off = mark;
  int* idx = (int*)sb.alloc((size_t)B * N1 * ns * sizeof(int));
  k_knn<8><<<dim3((B * N1 + 63) / 64), 64, 0, st>>>(pos2, pos1, idx, B, N2, N1);
  int rows = B * N1 * ns;
  int kpadG = pad32i(C2 + 3);
  h16* Xg = (h16*)sb.alloc((size_t)rows * kpadG * sizeof(h16));
  k_group_up<<<dim3((rows + 255) / 256), 256, 0, st>>>(pos1, pos2, f2, idx, Xg, B, N1, N2, C2, ns, kpadG);
  const h16* pooledSrc = Xg;
  int pooledC = C2 + 3;
  int pooledKpad = kpadG;
  if (nMlp > 0) {
    const h16* XhF = nullptr;
    run_mlp(st, sb, Xg, rows, kpadG, mlp, nMlp, nullptr, &XhF);
    pooledSrc = XhF;
    pooledC = mlp[nMlp - 1].cout;
    pooledKpad = pad32i(pooledC);
  }
  int P = B * N1;
  int cin2 = pooledC + C1;
  int kpad2 = pad32i(cin2);
  h16* X2 = (h16*)sb.alloc((size_t)P * kpad2 * sizeof(h16));
  long fn = (long)P * kpad2;
  k_fill_h<<<dim3((unsigned)((fn + 255) / 256)), 256, 0, st>>>(X2, fn);
  k_maxpool_h<<<dim3((P * pooledC + 255) / 256), 256, 0, st>>>(pooledSrc, X2, P, ns, pooledC, pooledKpad, kpad2, 0);
  if (C1 > 0 && f1)
    k_copyfeat_h<<<dim3((P * C1 + 255) / 256), 256, 0, st>>>(f1, X2, P, C1, kpad2, pooledC);
  run_mlp(st, sb, X2, P, kpad2, mlp2, nMlp2, out, nullptr);
}

static void fp_host(hipStream_t st, Bump& sb, size_t mark,
                    const float* pos1, const float* pos2, const float* f1w, const float* f2,
                    int B, int N1, int N2, int C2,
                    const Layer* fpc, int nL, const float* conv2W, const float* conv2b, float* dout)
{
  sb.off = mark;
  int rows = B * N1;
  int kpad = pad32i(C2 + 1);
  h16* Xh = (h16*)sb.alloc((size_t)rows * kpad * sizeof(h16));
  k_fp_interp<<<dim3((rows + 127) / 128), 128, 0, st>>>(pos1, pos2, f1w, f2, Xh, B, N1, N2, C2, kpad);
  const h16* XhF = nullptr;
  run_mlp(st, sb, Xh, rows, kpad, fpc, nL, nullptr, &XhF);
  int kpadc = pad32i(fpc[nL - 1].cout);   // 64
  h16* Wh = (h16*)sb.alloc((size_t)16 * kpadc * sizeof(h16));
  int wtotal = 16 * kpadc;
  k_convert_w<<<dim3((wtotal + 255) / 256), 256, 0, st>>>(conv2W, Wh, 3, fpc[nL - 1].cout, kpadc, wtotal);
  dim3 gg((rows + 63) / 64, 1);
  k_gemm_wmma<1><<<gg, 128, 0, st>>>(XhF, Wh, dout, rows, kpadc, 3);
  k_bias3<<<dim3((rows * 3 + 255) / 256), 256, 0, st>>>(dout, conv2b, rows);
}

// ------------------------------------------------------------------
// Entry point
// ------------------------------------------------------------------

extern "C" void kernel_launch(void* const* d_in, const int* in_sizes, int n_in,
                              void* d_out, int out_size, void* d_ws, size_t ws_size,
                              hipStream_t stream)
{
  (void)in_sizes; (void)n_in; (void)out_size; (void)ws_size;
  const int B = 2, N0 = 8192;

  const float* pc1 = (const float*)d_in[0];
  const float* w1  = (const float*)d_in[1];
  const float* pc2 = (const float*)d_in[2];
  const float* w2  = (const float*)d_in[3];
  // params flattened in sorted-key pytree order
  const float* bn1_b   = (const float*)d_in[4];
  const float* bn1_g   = (const float*)d_in[5];
  const float* conv1_W = (const float*)d_in[6];
  const float* conv2_W = (const float*)d_in[7];
  const float* conv2_b = (const float*)d_in[8];

  Layer fe[3], fp2[2], sa1[3], sa2[3], sa3[3], sa4[3];
  Layer su1m2[2], su2m[3], su2m2[1], su3m[3], su3m2[1];
  auto getL = [&](int base, int cin, const int* ch, int n, Layer* L) {
    for (int i = 0; i < n; ++i) {
      L[i].W = (const float*)d_in[base + 3 * i + 0];
      L[i].g = (const float*)d_in[base + 3 * i + 1];
      L[i].b = (const float*)d_in[base + 3 * i + 2];
      L[i].cin = cin; L[i].cout = ch[i];
      cin = ch[i];
    }
  };
  { const int c[3] = {64, 64, 64};    getL(9,  131, c, 3, fe);    }
  { const int c[2] = {128, 128};      getL(18, 129, c, 2, fp2);   }
  { const int c[3] = {16, 16, 32};    getL(24, 4,   c, 3, sa1);   }
  { const int c[3] = {32, 32, 64};    getL(33, 35,  c, 3, sa2);   }
  { const int c[3] = {64, 64, 128};   getL(42, 67,  c, 3, sa3);   }
  { const int c[3] = {128, 128, 256}; getL(51, 131, c, 3, sa4);   }
  { const int c[2] = {128, 128};      getL(60, 387, c, 2, su1m2); }
  { const int c[3] = {64, 64, 128};   getL(66, 131, c, 3, su2m);  }
  { const int c[1] = {128};           getL(75, 256, c, 1, su2m2); }
  { const int c[3] = {64, 64, 128};   getL(78, 131, c, 3, su3m);  }
  { const int c[1] = {128};           getL(87, 160, c, 1, su3m2); }

  Bump bump{ (char*)d_ws, 0 };
  // persistent level buffers
  float* l1p1  = (float*)bump.alloc((size_t)B * 4096 * 3   * 4);
  float* l1f1  = (float*)bump.alloc((size_t)B * 4096 * 32  * 4);
  float* l2p1  = (float*)bump.alloc((size_t)B * 1024 * 3   * 4);
  float* l2f1  = (float*)bump.alloc((size_t)B * 1024 * 64  * 4);
  float* l1p2  = (float*)bump.alloc((size_t)B * 4096 * 3   * 4);
  float* l1f2  = (float*)bump.alloc((size_t)B * 4096 * 32  * 4);
  float* l2p2  = (float*)bump.alloc((size_t)B * 1024 * 3   * 4);
  float* l2f2  = (float*)bump.alloc((size_t)B * 1024 * 64  * 4);
  float* l2f1n = (float*)bump.alloc((size_t)B * 1024 * 64  * 4);
  float* l3p1  = (float*)bump.alloc((size_t)B * 256  * 3   * 4);
  float* l3f1  = (float*)bump.alloc((size_t)B * 256  * 128 * 4);
  float* l4p1  = (float*)bump.alloc((size_t)B * 64   * 3   * 4);
  float* l4f1  = (float*)bump.alloc((size_t)B * 64   * 256 * 4);
  float* l3fn  = (float*)bump.alloc((size_t)B * 256  * 128 * 4);
  float* l2fn  = (float*)bump.alloc((size_t)B * 1024 * 128 * 4);
  float* l1fn  = (float*)bump.alloc((size_t)B * 4096 * 128 * 4);
  float* l2cat = (float*)bump.alloc((size_t)B * 1024 * 128 * 4);
  size_t mark = bump.off;

  // Set abstraction pyramid (src + tgt)
  sa_host(stream, bump, mark, pc1,  w1,   B, N0,   1,  4096, 0.02f, 16, sa1, 3, l1p1, l1f1);
  sa_host(stream, bump, mark, l1p1, l1f1, B, 4096, 32, 1024, 0.04f, 16, sa2, 3, l2p1, l2f1);
  sa_host(stream, bump, mark, pc2,  w2,   B, N0,   1,  4096, 0.02f, 16, sa1, 3, l1p2, l1f2);
  sa_host(stream, bump, mark, l1p2, l1f2, B, 4096, 32, 1024, 0.04f, 16, sa2, 3, l2p2, l2f2);

  // Flow embedding
  fe_host(stream, bump, mark, l2p1, l2p2, l2f1, l2f2, B, 1024, 64, fe, 3, l2f1n);

  // Deeper SA on embedded flow
  sa_host(stream, bump, mark, l2p1, l2f1n, B, 1024, 64,  256, 0.08f, 8, sa3, 3, l3p1, l3f1);
  sa_host(stream, bump, mark, l3p1, l3f1,  B, 256,  128, 64,  0.16f, 8, sa4, 3, l4p1, l4f1);

  // concat(l2f1, l2f1_new)
  k_copy_cols_f<<<dim3((B * 1024 * 64 + 255) / 256), 256, 0, stream>>>(l2f1,  l2cat, B * 1024, 64, 128, 0);
  k_copy_cols_f<<<dim3((B * 1024 * 64 + 255) / 256), 256, 0, stream>>>(l2f1n, l2cat, B * 1024, 64, 128, 64);

  // Up convolutions
  upconv_host(stream, bump, mark, l3p1, l4p1, l3f1,  l4f1, B, 256,  64,   128, 256, 8, nullptr, 0, su1m2, 2, l3fn);
  upconv_host(stream, bump, mark, l2p1, l3p1, l2cat, l3fn, B, 1024, 256,  128, 128, 8, su2m, 3, su2m2, 1, l2fn);
  upconv_host(stream, bump, mark, l1p1, l2p1, l1f1,  l2fn, B, 4096, 1024, 32,  128, 8, su3m, 3, su3m2, 1, l1fn);

  // Feature propagation + conv1 (as an extra bn-relu layer) + conv2 + bias
  Layer fpc[3] = { fp2[0], fp2[1], { conv1_W, bn1_g, bn1_b, 128, 64 } };
  fp_host(stream, bump, mark, pc1, l1p1, w1, l1fn, B, N0, 4096, 128,
          fpc, 3, conv2_W, conv2_b, (float*)d_out);
}
